// MambaStack_85057532330399
// MI455X (gfx1250) — compile-verified, matrix operand order checked by reading, not run on hardware
//
#include <hip/hip_runtime.h>
#include <hip/hip_bf16.h>

// ---------------- problem constants ----------------
#define D_MODEL 256
#define DI      512
#define NSTATE  16
#define DCONV   4
#define DTR     16
#define BSZ     32
#define TLEN    2048
#define BT      (BSZ * TLEN)          // 65536 rows
#define XDBLW   (DTR + 2 * NSTATE)    // 48

typedef __attribute__((ext_vector_type(16))) _Float16 v16h;
typedef __attribute__((ext_vector_type(8)))  _Float16 v8h;
typedef __attribute__((ext_vector_type(8)))  float    v8f;

// ---------------- workspace layout (bytes) ----------------
static constexpr size_t OFF_WIN  = 0;                               // 2*1024*256 f16
static constexpr size_t OFF_WX   = OFF_WIN  + 2ull*1024*256*2;      // 2*48*512 f16
static constexpr size_t OFF_WO   = OFF_WX   + 2ull*48*512*2;        // 2*256*512 f16
static constexpr size_t OFF_A    = OFF_WO   + 2ull*256*512*2;       // 2*512*16 f32
static constexpr size_t OFF_XN   = 2097152;                         // BT*256 f16  (32 MB)
static constexpr size_t OFF_XZ   = OFF_XN   + (size_t)BT*256*2;     // BT*1024 f16 (128 MB): [u_pre | z]
static constexpr size_t OFF_U    = OFF_XZ   + (size_t)BT*1024*2;    // BT*512 f16  (64 MB)
static constexpr size_t OFF_XDBL = OFF_U    + (size_t)BT*512*2;     // BT*48  f32  (12 MB)
static constexpr size_t OFF_DT   = OFF_XDBL + (size_t)BT*48*4;      // BT*512 f32  (128 MB)
static constexpr size_t OFF_Y    = OFF_DT   + (size_t)BT*512*4;     // BT*512 f16  (64 MB)

// ---------------- weight convert + A = -exp(A_log) ----------------
#define IN_N  (2*1024*256)
#define X_N   (2*48*512)
#define OUT_N (2*256*512)
#define A_N   (2*512*16)

__global__ void convert_kernel(const float* __restrict__ w_in, const float* __restrict__ w_x,
                               const float* __restrict__ w_out, const float* __restrict__ a_log,
                               _Float16* w16_in, _Float16* w16_x, _Float16* w16_out, float* Abuf) {
  int i = blockIdx.x * 256 + threadIdx.x;
  if (i < IN_N)  w16_in[i]  = (_Float16)w_in[i];
  if (i < X_N)   w16_x[i]   = (_Float16)w_x[i];
  if (i < OUT_N) w16_out[i] = (_Float16)w_out[i];
  if (i < A_N)   Abuf[i]    = -__expf(a_log[i]);
}

// ---------------- xcur = x * mask ----------------
__global__ void init_kernel(const float* __restrict__ x, const int* __restrict__ mask,
                            float* __restrict__ xcur) {
  size_t i = (size_t)blockIdx.x * 256 + threadIdx.x;   // BT*256 elements
  float m = mask[i >> 8] ? 1.f : 0.f;
  xcur[i] = x[i] * m;
}

// ---------------- LayerNorm -> f16 (one wave32 per 256-wide row) ----------------
__global__ void ln_kernel(const float* __restrict__ x, const float* __restrict__ w,
                          const float* __restrict__ b, _Float16* __restrict__ out) {
  int row  = blockIdx.x * 8 + (threadIdx.x >> 5);
  int lane = threadIdx.x & 31;
  const float* xr = x + (size_t)row * D_MODEL + lane * 8;
  float4 p0 = *(const float4*)xr;
  float4 p1 = *(const float4*)(xr + 4);
  float v[8] = {p0.x, p0.y, p0.z, p0.w, p1.x, p1.y, p1.z, p1.w};
  float s = 0.f, ss = 0.f;
  #pragma unroll
  for (int j = 0; j < 8; ++j) { s += v[j]; ss += v[j] * v[j]; }
  #pragma unroll
  for (int off = 16; off > 0; off >>= 1) {
    s  += __shfl_xor(s,  off, 32);
    ss += __shfl_xor(ss, off, 32);
  }
  float mu  = s * (1.f / D_MODEL);
  float var = ss * (1.f / D_MODEL) - mu * mu;
  float rs  = rsqrtf(var + 1e-5f);
  v8h o;
  #pragma unroll
  for (int j = 0; j < 8; ++j)
    o[j] = (_Float16)((v[j] - mu) * rs * w[lane * 8 + j] + b[lane * 8 + j]);
  *(v8h*)(out + (size_t)row * D_MODEL + lane * 8) = o;
}

// ------- WMMA: 1 m-tile x NT n-tiles per wave; A fragment reused NT times --------
// A: MxK row-major f16.  W: NxK row-major f16 (so W rows are B columns).
// Per K-step: batch-load A + all NT B fragments, then issue the WMMA group, so
// waits can be partial (loadcnt N-1, N-2, ...) and WMMAs overlap trailing loads.
template <int K, int NT>
__device__ inline void wmma_tileN(const _Float16* __restrict__ A,
                                  const _Float16* __restrict__ W,
                                  int m0, int n0, v8f acc[NT]) {
  int lane = threadIdx.x & 31;
  int r  = lane & 15;        // A row / B col served by this lane
  int kh = lane >> 4;        // K-half select (documented 16-bit A/B layout)
  const _Float16* arow = A + (size_t)(m0 + r) * K + kh * 8;
  const _Float16* wrow = W + (size_t)(n0 + r) * K + kh * 8;
  #pragma unroll 2
  for (int k0 = 0; k0 < K; k0 += 32) {
    v8h alo = *(const v8h*)(arow + k0);
    v8h ahi = *(const v8h*)(arow + k0 + 16);
    v16h bmv[NT];
    #pragma unroll
    for (int j = 0; j < NT; ++j) {
      v8h blo = *(const v8h*)(wrow + (size_t)j * 16 * K + k0);
      v8h bhi = *(const v8h*)(wrow + (size_t)j * 16 * K + k0 + 16);
      #pragma unroll
      for (int e = 0; e < 8; ++e) { bmv[j][e] = blo[e]; bmv[j][8 + e] = bhi[e]; }
    }
    v16h a;
    #pragma unroll
    for (int e = 0; e < 8; ++e) { a[e] = alo[e]; a[8 + e] = ahi[e]; }
    #pragma unroll
    for (int j = 0; j < NT; ++j)
      acc[j] = __builtin_amdgcn_wmma_f32_16x16x32_f16(false, a, false, bmv[j],
                                                      (short)0, acc[j], false, false);
  }
}

// ---------------- GEMM1: xz = xn @ in_proj_w^T  (N=1024, contiguous store) -------
__global__ void gemm_in_kernel(const _Float16* __restrict__ A, const _Float16* __restrict__ W,
                               _Float16* __restrict__ xz) {
  int job = blockIdx.x * 8 + (threadIdx.x >> 5);    // 4096 m-tiles * 16 n-groups
  int mt = job >> 4, ng = job & 15;
  v8f acc[4] = {};
  wmma_tileN<256, 4>(A, W, mt * 16, ng * 64, acc);
  int lane = threadIdx.x & 31;
  int kh = lane >> 4;
  #pragma unroll
  for (int j = 0; j < 4; ++j) {
    int col = ng * 64 + j * 16 + (lane & 15);
    #pragma unroll
    for (int rI = 0; rI < 8; ++rI) {
      int row = mt * 16 + rI + 8 * kh;
      xz[(size_t)row * 1024 + col] = (_Float16)acc[j][rI];
    }
  }
}

// ---------------- causal depthwise conv (k=4) + SiLU  (reads xz[:, :512]) --------
__global__ void conv_silu_kernel(const _Float16* __restrict__ xz,
                                 const float* __restrict__ cw, const float* __restrict__ cb,
                                 _Float16* __restrict__ u) {
  size_t i = (size_t)blockIdx.x * 256 + threadIdx.x; // BT*DI threads
  int d = (int)(i & (DI - 1));
  size_t row = i >> 9;
  int t = (int)(row & (TLEN - 1));
  float acc = cb[d];
  #pragma unroll
  for (int o = 0; o < DCONV; ++o)
    if (t - o >= 0)
      acc += (float)xz[(row - o) * 1024 + d] * cw[d * DCONV + (DCONV - 1 - o)];
  float sv = acc / (1.f + __expf(-acc));             // SiLU
  u[row * DI + d] = (_Float16)sv;
}

// ---------------- GEMM2: x_dbl = u @ x_proj_w^T  (N=48 -> NT=3) ------------------
__global__ void gemm_x_kernel(const _Float16* __restrict__ A, const _Float16* __restrict__ W,
                              float* __restrict__ xdbl) {
  int mt = blockIdx.x * 8 + (threadIdx.x >> 5);      // 4096 m-tiles
  v8f acc[3] = {};
  wmma_tileN<512, 3>(A, W, mt * 16, 0, acc);
  int lane = threadIdx.x & 31;
  int kh = lane >> 4;
  #pragma unroll
  for (int j = 0; j < 3; ++j) {
    int col = j * 16 + (lane & 15);
    #pragma unroll
    for (int rI = 0; rI < 8; ++rI) {
      int row = mt * 16 + rI + 8 * kh;
      xdbl[(size_t)row * XDBLW + col] = acc[j][rI];
    }
  }
}

// ---------------- dt = softplus(dt_in @ dt_proj_w^T + b) (K=16, VALU) ------------
__global__ void dtproj_kernel(const float* __restrict__ xdbl, const float* __restrict__ w,
                              const float* __restrict__ b, float* __restrict__ dt) {
  size_t i = (size_t)blockIdx.x * 256 + threadIdx.x; // BT*DI
  int d = (int)(i & (DI - 1));
  size_t row = i >> 9;
  const float* xr = xdbl + row * XDBLW;
  float s = b[d];
  #pragma unroll
  for (int r = 0; r < DTR; ++r) s += xr[r] * w[d * DTR + r];
  float sp = (s > 20.f) ? s : log1pf(__expf(s));
  dt[row * DI + d] = sp;
}

// ------- selective scan: quad of lanes per (b,d); 4 states/lane; pipelined -------
__global__ void scan_kernel(const float* __restrict__ dt, const _Float16* __restrict__ u,
                            const _Float16* __restrict__ xz, const float* __restrict__ xdbl,
                            const float* __restrict__ A, const float* __restrict__ skipD,
                            _Float16* __restrict__ y) {
  int tid = blockIdx.x * 256 + threadIdx.x;          // B*DI*4 = 65536 lanes
  int q = tid & 3;                                   // state quarter
  int d = (tid >> 2) & (DI - 1);
  int b = tid >> 11;
  float Ar[4];
  #pragma unroll
  for (int n = 0; n < 4; ++n) Ar[n] = A[d * NSTATE + q * 4 + n];
  float sD = skipD[d];
  float h[4] = {0.f, 0.f, 0.f, 0.f};

  size_t rowbase = (size_t)b * TLEN;
  const float*    dtp = dt + rowbase * DI + d;
  const _Float16* up  = u  + rowbase * DI + d;
  const _Float16* zp  = xz + rowbase * 1024 + DI + d;          // z half of xz
  const float*    xp  = xdbl + rowbase * XDBLW + DTR + q * 4;  // B quarter; C at +32

  // software pipeline: preload t = 0
  float  dtv = dtp[0];
  float  uv  = (float)up[0];
  float  zv  = (float)zp[0];
  float4 Bq  = *(const float4*)(xp);
  float4 Cq  = *(const float4*)(xp + 2 * NSTATE);

  for (int t = 0; t < TLEN; ++t) {
    // issue next-step loads before the serial exp/fma chain
    float dtn = 0.f, un = 0.f, zn = 0.f;
    float4 Bn = {0.f, 0.f, 0.f, 0.f}, Cn = {0.f, 0.f, 0.f, 0.f};
    if (t + 1 < TLEN) {
      dtn = dtp[(size_t)(t + 1) * DI];
      un  = (float)up[(size_t)(t + 1) * DI];
      zn  = (float)zp[(size_t)(t + 1) * 1024];
      Bn  = *(const float4*)(xp + (size_t)(t + 1) * XDBLW);
      Cn  = *(const float4*)(xp + (size_t)(t + 1) * XDBLW + 2 * NSTATE);
    }
    if (t + 8 < TLEN) {                               // global_prefetch_b8
      __builtin_prefetch(dtp + (size_t)(t + 8) * DI, 0, 0);
      __builtin_prefetch(xp + (size_t)(t + 8) * XDBLW, 0, 0);
    }
    float Bv[4] = {Bq.x, Bq.y, Bq.z, Bq.w};
    float Cv[4] = {Cq.x, Cq.y, Cq.z, Cq.w};
    float dtu = dtv * uv;
    float yv = 0.f;
    #pragma unroll
    for (int n = 0; n < 4; ++n) {
      float dA = __expf(dtv * Ar[n]);
      h[n] = dA * h[n] + dtu * Bv[n];
      yv += h[n] * Cv[n];
    }
    yv += __shfl_xor(yv, 1, 32);                      // reduce quad partial sums
    yv += __shfl_xor(yv, 2, 32);
    if (q == 0) {
      yv += uv * sD;
      yv *= zv / (1.f + __expf(-zv));                 // * silu(z)
      y[(rowbase + t) * DI + d] = (_Float16)yv;
    }
    dtv = dtn; uv = un; zv = zn; Bq = Bn; Cq = Cn;
  }
}

// ---------------- GEMM3: h = y @ out_proj_w^T ; xcur = (xcur + h) * mask ---------
__global__ void gemm_out_kernel(const _Float16* __restrict__ A, const _Float16* __restrict__ W,
                                const int* __restrict__ mask, float* __restrict__ xcur) {
  int job = blockIdx.x * 8 + (threadIdx.x >> 5);     // 4096 m-tiles * 4 n-groups
  int mt = job >> 2, ng = job & 3;
  v8f acc[4] = {};
  wmma_tileN<512, 4>(A, W, mt * 16, ng * 64, acc);
  int lane = threadIdx.x & 31;
  int kh = lane >> 4;
  #pragma unroll
  for (int j = 0; j < 4; ++j) {
    int col = ng * 64 + j * 16 + (lane & 15);
    #pragma unroll
    for (int rI = 0; rI < 8; ++rI) {
      int row = mt * 16 + rI + 8 * kh;
      float m = mask[row] ? 1.f : 0.f;
      size_t idx = (size_t)row * D_MODEL + col;
      xcur[idx] = (xcur[idx] + acc[j][rI]) * m;
    }
  }
}

// ---------------- host launch ----------------
extern "C" void kernel_launch(void* const* d_in, const int* in_sizes, int n_in,
                              void* d_out, int out_size, void* d_ws, size_t ws_size,
                              hipStream_t stream) {
  const float* x       = (const float*)d_in[0];
  const int*   mask    = (const int*)d_in[1];
  const float* ln_w    = (const float*)d_in[2];
  const float* ln_b    = (const float*)d_in[3];
  const float* in_w    = (const float*)d_in[4];
  const float* conv_w  = (const float*)d_in[5];
  const float* conv_b  = (const float*)d_in[6];
  const float* xp_w    = (const float*)d_in[7];
  const float* dtp_w   = (const float*)d_in[8];
  const float* dtp_b   = (const float*)d_in[9];
  const float* a_log   = (const float*)d_in[10];
  const float* skip_D  = (const float*)d_in[11];
  const float* out_w   = (const float*)d_in[12];

  char* ws = (char*)d_ws;
  _Float16* w16_in  = (_Float16*)(ws + OFF_WIN);
  _Float16* w16_x   = (_Float16*)(ws + OFF_WX);
  _Float16* w16_out = (_Float16*)(ws + OFF_WO);
  float*    Abuf    = (float*)(ws + OFF_A);
  _Float16* xn16    = (_Float16*)(ws + OFF_XN);
  _Float16* xz16    = (_Float16*)(ws + OFF_XZ);
  _Float16* u16     = (_Float16*)(ws + OFF_U);
  float*    xdbl    = (float*)(ws + OFF_XDBL);
  float*    dtb     = (float*)(ws + OFF_DT);
  _Float16* y16     = (_Float16*)(ws + OFF_Y);
  float*    xcur    = (float*)d_out;

  convert_kernel<<<(IN_N + 255) / 256, 256, 0, stream>>>(in_w, xp_w, out_w, a_log,
                                                         w16_in, w16_x, w16_out, Abuf);
  init_kernel<<<(BT * D_MODEL) / 256, 256, 0, stream>>>(x, mask, xcur);

  for (int l = 0; l < 2; ++l) {
    ln_kernel<<<BT / 8, 256, 0, stream>>>(xcur, ln_w + l * D_MODEL, ln_b + l * D_MODEL, xn16);
    gemm_in_kernel<<<(BT / 16) * 16 / 8, 256, 0, stream>>>(
        xn16, w16_in + (size_t)l * 1024 * D_MODEL, xz16);
    conv_silu_kernel<<<(BT * DI) / 256, 256, 0, stream>>>(
        xz16, conv_w + l * DI * DCONV, conv_b + l * DI, u16);
    gemm_x_kernel<<<(BT / 16) / 8, 256, 0, stream>>>(
        u16, w16_x + (size_t)l * XDBLW * DI, xdbl);
    dtproj_kernel<<<(BT * DI) / 256, 256, 0, stream>>>(
        xdbl, dtp_w + l * DI * DTR, dtp_b + l * DI, dtb);
    scan_kernel<<<(BSZ * DI * 4) / 256, 256, 0, stream>>>(
        dtb, u16, xz16, xdbl, Abuf + l * DI * NSTATE, skip_D + l * DI, y16);
    gemm_out_kernel<<<(BT / 16) * 4 / 8, 256, 0, stream>>>(
        y16, w16_out + (size_t)l * D_MODEL * DI, mask, xcur);
  }
}
